// PtrExtractorRLStop_32126355374081
// MI455X (gfx1250) — compile-verified, matrix operand order checked by reading, not run on hardware
//
#include <hip/hip_runtime.h>
#include <hip/hip_bf16.h>

#define N_MEM   50000
#define NP1     50001
#define PADROWS 50016
#define D       512
#define H       512
#define TSTEPS  16
#define STOPIDX 50000
#define NEGVAL  (-1e18f)
#define NW      2048          // waves in the big scans (256 blocks x 256 thr)
#define MT      3126          // ceil(50001/16) M-tiles
#define KS      16            // k-split for the small GEMVs

typedef __attribute__((ext_vector_type(16))) __bf16 v16bf;
typedef __attribute__((ext_vector_type(8)))  float  v8f;

__device__ __forceinline__ unsigned short f2bf(float f) {
    unsigned u = __builtin_bit_cast(unsigned, f);
    unsigned r = (u + 0x7fffu + ((u >> 16) & 1u)) >> 16;
    return (unsigned short)r;
}
__device__ __forceinline__ float bf2f(unsigned short v) {
    unsigned u = ((unsigned)v) << 16;
    return __builtin_bit_cast(float, u);
}
__device__ __forceinline__ float sigf(float x) { return 1.0f / (1.0f + __expf(-x)); }

// ---------------------------------------------------------------- init
__global__ void pn_init(const float* ih, const float* ic, const float* ii,
                        float* h, float* c, float* x, int* sel, int* donep) {
    int t = blockIdx.x * blockDim.x + threadIdx.x;
    if (t < H) { h[t] = ih[t]; c[t] = ic[t]; }
    if (t < D) { x[t] = ii[t]; }
    if (t < NP1) sel[t] = 0;
    if (t == 0) *donep = 0;
}

// --------------------------------------------------- feature GEMM (WMMA)
// attn_feat = mem @ attn_wm ; hop_feat = mem @ hop_wm ; mem = [attn_mem; stop]
// one wave per 16x16 output tile; K loop in steps of 32 (bf16 WMMA)
__global__ void pn_feat_gemm(const float* __restrict__ attn_mem,
                             const float* __restrict__ stop,
                             const float* __restrict__ attn_wm,
                             const float* __restrict__ hop_wm,
                             unsigned short* __restrict__ attn_feat,
                             unsigned short* __restrict__ hop_feat) {
    const int l    = threadIdx.x;        // 0..31 (one wave per threadIdx.y)
    const int half = l >> 4;             // 0 or 1
    const int lm   = l & 15;
    const int m0   = blockIdx.x * 16;
    const int n0   = (blockIdx.y * blockDim.y + threadIdx.y) * 16;

    int rA = m0 + lm;
    if (rA >= NP1) rA = STOPIDX;                       // clamp; masked on store
    const float* arow = (rA < N_MEM) ? (attn_mem + (size_t)rA * D) : stop;
    const int kA   = half * 8;                         // lane K sub-offset
    const int colB = n0 + lm;

    v8f accA = {0.f,0.f,0.f,0.f,0.f,0.f,0.f,0.f};
    v8f accH = {0.f,0.f,0.f,0.f,0.f,0.f,0.f,0.f};

#pragma unroll 2
    for (int kb = 0; kb < D; kb += 32) {
        v16bf a, bA, bH;
#pragma unroll
        for (int i = 0; i < 8; ++i) {
            a[i]     = (__bf16)arow[kb + kA + i];
            a[8 + i] = (__bf16)arow[kb + 16 + kA + i];
            bA[i]     = (__bf16)attn_wm[(size_t)(kb + kA + i)      * H + colB];
            bA[8 + i] = (__bf16)attn_wm[(size_t)(kb + 16 + kA + i) * H + colB];
            bH[i]     = (__bf16)hop_wm[(size_t)(kb + kA + i)      * H + colB];
            bH[8 + i] = (__bf16)hop_wm[(size_t)(kb + 16 + kA + i) * H + colB];
        }
        accA = __builtin_amdgcn_wmma_f32_16x16x32_bf16(false, a, false, bA,
                                                       (short)0, accA, false, false);
        accH = __builtin_amdgcn_wmma_f32_16x16x32_bf16(false, a, false, bH,
                                                       (short)0, accH, false, false);
    }
#pragma unroll
    for (int r = 0; r < 8; ++r) {
        int row = m0 + r + 8 * half;
        if (row < NP1) {
            attn_feat[(size_t)row * H + n0 + lm] = f2bf(accA[r]);
            hop_feat [(size_t)row * H + n0 + lm] = f2bf(accH[r]);
        }
    }
}

// ---------------------------------------------------------------- LSTM
__global__ void pn_lstm_gates(const float* __restrict__ w_ih,
                              const float* __restrict__ w_hh,
                              const float* __restrict__ b_ih,
                              const float* __restrict__ b_hh,
                              const float* __restrict__ x,
                              const float* __restrict__ h,
                              float* __restrict__ gates) {
    int gt   = blockIdx.x * blockDim.x + threadIdx.x;
    int wid  = gt >> 5;                   // 0..2047
    int lane = gt & 31;
    const float* wi = w_ih + (size_t)wid * D;
    const float* wh = w_hh + (size_t)wid * H;
    float s = 0.f;
    for (int k = lane; k < D; k += 32) s += wi[k] * x[k];
    for (int k = lane; k < H; k += 32) s += wh[k] * h[k];
#pragma unroll
    for (int off = 16; off; off >>= 1) s += __shfl_xor(s, off, 32);
    if (lane == 0) gates[wid] = s + b_ih[wid] + b_hh[wid];
}

__global__ void pn_lstm_update(const float* __restrict__ gates,
                               float* __restrict__ h, float* __restrict__ c) {
    int j = blockIdx.x * blockDim.x + threadIdx.x;   // 0..511
    float ig = sigf(gates[j]);
    float fg = sigf(gates[H + j]);
    float gg = tanhf(gates[2 * H + j]);
    float og = sigf(gates[3 * H + j]);
    float cn = fg * c[j] + ig * gg;
    c[j] = cn;
    h[j] = og * tanhf(cn);
}

// ---------------------------------------------------------------- GEMV
// k-split partials: part[ks][n] = sum_{k in chunk ks} q[k]*W[k*H+n]
__global__ void pn_gemv_part(const float* __restrict__ q,
                             const float* __restrict__ W,
                             float* __restrict__ part) {
    int n  = blockIdx.x * blockDim.x + threadIdx.x;  // 0..511 (grid.x = 2)
    int ks = blockIdx.y;                             // 0..KS-1
    int k0 = ks * (H / KS);
    float s = 0.f;
#pragma unroll
    for (int k = k0; k < k0 + H / KS; ++k) s += q[k] * W[(size_t)k * H + n];
    part[(size_t)ks * H + n] = s;
}

__global__ void pn_gemv_reduce(const float* __restrict__ part,
                               float* __restrict__ out) {
    int n = blockIdx.x * blockDim.x + threadIdx.x;
    float s = 0.f;
#pragma unroll
    for (int ks = 0; ks < KS; ++ks) s += part[(size_t)ks * H + n];
    out[n] = s;
}

// ------------------------------------------------------- glimpse scan
__global__ void pn_glimpse_scan(const unsigned short* __restrict__ feat,
                                const float* __restrict__ qh,
                                const float* __restrict__ vw,
                                float* __restrict__ pmax,
                                float* __restrict__ psum,
                                float* __restrict__ pvec) {
    int gt   = blockIdx.x * blockDim.x + threadIdx.x;
    int wid  = gt >> 5;
    int lane = gt & 31;
    int h0   = lane * 16;
    float qv[16], vv[16];
#pragma unroll
    for (int i = 0; i < 16; ++i) { qv[i] = qh[h0 + i]; vv[i] = vw[h0 + i]; }

    float m = -INFINITY, Z = 0.f;
    float acc[16];
#pragma unroll
    for (int i = 0; i < 16; ++i) acc[i] = 0.f;

    for (int row = wid; row < NP1; row += NW) {
        const unsigned short* fr = feat + (size_t)row * H + h0;
        __builtin_prefetch(fr + (size_t)NW * H, 0, 1);
        uint4 pa = *(const uint4*)fr;
        uint4 pb = *(const uint4*)(fr + 8);
        unsigned wbits[8] = {pa.x, pa.y, pa.z, pa.w, pb.x, pb.y, pb.z, pb.w};
        float f[16];
        float sp = 0.f;
#pragma unroll
        for (int i = 0; i < 8; ++i) {
            f[2*i]   = bf2f((unsigned short)(wbits[i] & 0xffffu));
            f[2*i+1] = bf2f((unsigned short)(wbits[i] >> 16));
        }
#pragma unroll
        for (int i = 0; i < 16; ++i) sp += tanhf(f[i] + qv[i]) * vv[i];
#pragma unroll
        for (int off = 16; off; off >>= 1) sp += __shfl_xor(sp, off, 32);
        if (sp > m) {
            float r = __expf(m - sp);
            Z = Z * r + 1.0f;
#pragma unroll
            for (int i = 0; i < 16; ++i) acc[i] = acc[i] * r + f[i];
            m = sp;
        } else {
            float w = __expf(sp - m);
            Z += w;
#pragma unroll
            for (int i = 0; i < 16; ++i) acc[i] += w * f[i];
        }
    }
    if (lane == 0) { pmax[wid] = m; psum[wid] = Z; }
#pragma unroll
    for (int i = 0; i < 16; ++i) pvec[(size_t)wid * H + h0 + i] = acc[i];
}

// global softmax stats over wave partials: MZ[0]=max, MZ[1]=sumexp
__global__ void pn_softmax_stats(const float* __restrict__ pmax,
                                 const float* __restrict__ psum,
                                 float* __restrict__ MZ) {
    __shared__ float red[512];
    int j = threadIdx.x;
    float lm = -INFINITY;
    for (int w = j; w < NW; w += 512) lm = fmaxf(lm, pmax[w]);
    red[j] = lm; __syncthreads();
    for (int st = 256; st; st >>= 1) {
        if (j < st) red[j] = fmaxf(red[j], red[j + st]);
        __syncthreads();
    }
    float M = red[0];
    __syncthreads();
    float z = 0.f;
    for (int w = j; w < NW; w += 512) z += psum[w] * __expf(pmax[w] - M);
    red[j] = z; __syncthreads();
    for (int st = 256; st; st >>= 1) {
        if (j < st) red[j] += red[j + st];
        __syncthreads();
    }
    if (j == 0) { MZ[0] = M; MZ[1] = red[0]; }
}

// column-parallel weighted combine: qout[j] = sum_w pvec[w][j]*exp(pmax[w]-M)/Z
__global__ void pn_glimpse_vec(const float* __restrict__ pvec,
                               const float* __restrict__ pmax,
                               const float* __restrict__ MZ,
                               float* __restrict__ qout) {
    int j = blockIdx.x * blockDim.x + threadIdx.x;   // 0..511 (16 blocks x 32)
    float M = MZ[0], Z = MZ[1];
    float a = 0.f;
    for (int w = 0; w < NW; ++w) a += pvec[(size_t)w * H + j] * __expf(pmax[w] - M);
    qout[j] = a / Z;
}

// ------------------------------------------------------- pointer scan
__global__ void pn_ptr_scan(const unsigned short* __restrict__ feat,
                            const float* __restrict__ qa,
                            const float* __restrict__ vw,
                            const int* __restrict__ sel,
                            const float* __restrict__ gmb,
                            float* __restrict__ pbv, int* __restrict__ pbi,
                            float* __restrict__ pbs,
                            float* __restrict__ psm, float* __restrict__ pss) {
    int gt   = blockIdx.x * blockDim.x + threadIdx.x;
    int wid  = gt >> 5;
    int lane = gt & 31;
    int h0   = lane * 16;
    float qv[16], vv[16];
#pragma unroll
    for (int i = 0; i < 16; ++i) { qv[i] = qa[h0 + i]; vv[i] = vw[h0 + i]; }

    float m = -INFINITY, Z = 0.f;
    float bestv = -INFINITY, bests = 0.f;
    int   besti = 0x7fffffff;

    for (int row = wid; row < NP1; row += NW) {
        const unsigned short* fr = feat + (size_t)row * H + h0;
        __builtin_prefetch(fr + (size_t)NW * H, 0, 1);
        uint4 pa = *(const uint4*)fr;
        uint4 pb = *(const uint4*)(fr + 8);
        unsigned wbits[8] = {pa.x, pa.y, pa.z, pa.w, pb.x, pb.y, pb.z, pb.w};
        float sp = 0.f;
#pragma unroll
        for (int i = 0; i < 8; ++i) {
            float f0 = bf2f((unsigned short)(wbits[i] & 0xffffu));
            float f1 = bf2f((unsigned short)(wbits[i] >> 16));
            sp += tanhf(f0 + qv[2*i])     * vv[2*i];
            sp += tanhf(f1 + qv[2*i + 1]) * vv[2*i + 1];
        }
#pragma unroll
        for (int off = 16; off; off >>= 1) sp += __shfl_xor(sp, off, 32);
        float s = sel[row] ? NEGVAL : sp;          // mask selected
        float sg = s + gmb[row];
        if (sg > bestv || (sg == bestv && row < besti)) {
            bestv = sg; besti = row; bests = s;
        }
        if (s > m) { Z = Z * __expf(m - s) + 1.0f; m = s; }
        else       { Z += __expf(s - m); }
    }
    if (lane == 0) {
        pbv[wid] = bestv; pbi[wid] = besti; pbs[wid] = bests;
        psm[wid] = m;     pss[wid] = Z;
    }
}

__global__ void pn_ptr_combine(const float* __restrict__ pbv,
                               const int* __restrict__ pbi,
                               const float* __restrict__ pbs,
                               const float* __restrict__ psm,
                               const float* __restrict__ pss,
                               const float* __restrict__ attn_mem,
                               int* __restrict__ sel, int* __restrict__ donep,
                               float* __restrict__ x,
                               int* __restrict__ out_i, float* __restrict__ out_f,
                               int t) {
    __shared__ float sv[512]; __shared__ int si[512];
    __shared__ float ss[512]; __shared__ float sm[512];
    __shared__ int sh_out, sh_nd;
    int j = threadIdx.x;
    float bv = -INFINITY, bs = 0.f, lm = -INFINITY;
    int bi = 0x7fffffff;
    for (int w = j; w < NW; w += 512) {
        float v = pbv[w]; int i = pbi[w];
        if (v > bv || (v == bv && i < bi)) { bv = v; bi = i; bs = pbs[w]; }
        lm = fmaxf(lm, psm[w]);
    }
    sv[j] = bv; si[j] = bi; ss[j] = bs; sm[j] = lm;
    __syncthreads();
    for (int st = 256; st; st >>= 1) {
        if (j < st) {
            if (sv[j + st] > sv[j] || (sv[j + st] == sv[j] && si[j + st] < si[j])) {
                sv[j] = sv[j + st]; si[j] = si[j + st]; ss[j] = ss[j + st];
            }
            sm[j] = fmaxf(sm[j], sm[j + st]);
        }
        __syncthreads();
    }
    float M = sm[0]; int out = si[0]; float bests = ss[0];
    __syncthreads();
    float z = 0.f;
    for (int w = j; w < NW; w += 512) z += pss[w] * __expf(psm[w] - M);
    sv[j] = z; __syncthreads();
    for (int st = 256; st; st >>= 1) {
        if (j < st) sv[j] += sv[j + st];
        __syncthreads();
    }
    if (j == 0) {
        int done  = *donep;
        float Zt  = sv[0];
        float logp = bests - (M + __logf(Zt));
        out_i[t]          = done ? STOPIDX : out;
        out_f[TSTEPS + t] = done ? 0.f : logp;
        if (!done) sel[out] = 1;
        int nd = (done || out == STOPIDX) ? 1 : 0;
        *donep = nd;
        sh_out = out; sh_nd = nd;
    }
    __syncthreads();
    if (!sh_nd) x[j] = attn_mem[(size_t)sh_out * D + j];
}

// ---------------------------------------------------------------- host
extern "C" void kernel_launch(void* const* d_in, const int* in_sizes, int n_in,
                              void* d_out, int out_size, void* d_ws, size_t ws_size,
                              hipStream_t stream) {
    (void)in_sizes; (void)n_in; (void)out_size; (void)ws_size;
    const float* attn_mem = (const float*)d_in[0];
    const float* stop     = (const float*)d_in[1];
    const float* init_h   = (const float*)d_in[2];
    const float* init_c   = (const float*)d_in[3];
    const float* init_i   = (const float*)d_in[4];
    const float* attn_wm  = (const float*)d_in[5];
    const float* attn_wq  = (const float*)d_in[6];
    const float* attn_v   = (const float*)d_in[7];
    const float* hop_wm   = (const float*)d_in[8];
    const float* hop_wq   = (const float*)d_in[9];
    const float* hop_v    = (const float*)d_in[10];
    const float* w_ih     = (const float*)d_in[11];
    const float* w_hh     = (const float*)d_in[12];
    const float* b_ih     = (const float*)d_in[13];
    const float* b_hh     = (const float*)d_in[14];
    const float* gumbel   = (const float*)d_in[15];

    // ---- carve workspace
    char* w = (char*)d_ws;
    unsigned short* attn_feat = (unsigned short*)w;                // PADROWS*H bf16
    unsigned short* hop_feat  = attn_feat + (size_t)PADROWS * H;
    float* fb = (float*)(hop_feat + (size_t)PADROWS * H);
    float* h_s   = fb;            fb += H;
    float* c_s   = fb;            fb += H;
    float* x_s   = fb;            fb += D;
    float* qh_s  = fb;            fb += H;
    float* qv_s  = fb;            fb += H;
    float* qa_s  = fb;            fb += H;
    float* gates = fb;            fb += 4 * H;
    float* gpart = fb;            fb += (size_t)KS * H;
    float* MZ    = fb;            fb += 2;
    float* gpmax = fb;            fb += NW;
    float* gpsum = fb;            fb += NW;
    float* gpvec = fb;            fb += (size_t)NW * H;
    float* ppbv  = fb;            fb += NW;
    float* ppbs  = fb;            fb += NW;
    float* ppsm  = fb;            fb += NW;
    float* ppss  = fb;            fb += NW;
    int*   ppbi  = (int*)fb;      fb += NW;
    int*   sel   = (int*)fb;      fb += PADROWS;
    int*   donep = (int*)fb;      fb += 1;

    int* out_i   = (int*)d_out;
    float* out_f = (float*)d_out;

    pn_init<<<(NP1 + 255) / 256, 256, 0, stream>>>(init_h, init_c, init_i,
                                                   h_s, c_s, x_s, sel, donep);

    pn_feat_gemm<<<dim3(MT, 8), dim3(32, 4), 0, stream>>>(
        attn_mem, stop, attn_wm, hop_wm, attn_feat, hop_feat);

    for (int t = 0; t < TSTEPS; ++t) {
        pn_lstm_gates<<<256, 256, 0, stream>>>(w_ih, w_hh, b_ih, b_hh,
                                               x_s, h_s, gates);
        pn_lstm_update<<<2, 256, 0, stream>>>(gates, h_s, c_s);

        pn_gemv_part<<<dim3(2, KS), 256, 0, stream>>>(h_s, hop_wq, gpart);
        pn_gemv_reduce<<<2, 256, 0, stream>>>(gpart, qh_s);

        pn_glimpse_scan<<<256, 256, 0, stream>>>(hop_feat, qh_s, hop_v,
                                                 gpmax, gpsum, gpvec);
        pn_softmax_stats<<<1, 512, 0, stream>>>(gpmax, gpsum, MZ);
        pn_glimpse_vec<<<16, 32, 0, stream>>>(gpvec, gpmax, MZ, qv_s);

        pn_gemv_part<<<dim3(2, KS), 256, 0, stream>>>(qv_s, attn_wq, gpart);
        pn_gemv_reduce<<<2, 256, 0, stream>>>(gpart, qa_s);

        pn_ptr_scan<<<256, 256, 0, stream>>>(attn_feat, qa_s, attn_v, sel,
                                             gumbel + (size_t)t * NP1,
                                             ppbv, ppbi, ppbs, ppsm, ppss);
        pn_ptr_combine<<<1, 512, 0, stream>>>(ppbv, ppbi, ppbs, ppsm, ppss,
                                              attn_mem, sel, donep, x_s,
                                              out_i, out_f, t);
    }
}
